// GptOssTopKRouter_57354993271422
// MI455X (gfx1250) — matrix operand order repaired and checked by reading, not compiled, not run on hardware
//
#include <hip/hip_runtime.h>
#include <math.h>

// ---------------------------------------------------------------------------
// GPT-OSS top-k router for MI455X (gfx1250, wave32, WMMA).
//   hidden [16384, 2880] f32, weight [128, 2880] f32, bias [128] f32
//   logits = hidden @ weight^T + bias ; top4 ; softmax(top4)
// Strategy: bf16 WMMA (f32 accum) -> HBM-bound (~8us floor @ 23.3 TB/s).
// ---------------------------------------------------------------------------

typedef __attribute__((ext_vector_type(16))) __bf16 v16bf;
typedef __attribute__((ext_vector_type(8)))  float  v8f;
typedef __attribute__((ext_vector_type(4)))  float  v4f;

#define T_TOK  16384
#define H_DIM  2880
#define E_DIM  128
#define BLK_M  128     // tokens per workgroup
#define KSTEP  32      // K per bf16 WMMA
#define TOPK   4

// Load two contiguous groups of 8 fp32 and pack to 16 bf16 (WMMA reg layout).
__device__ __forceinline__ v16bf load_cvt_bf16(const float* __restrict__ p0,
                                               const float* __restrict__ p1) {
    v4f x0 = *(const v4f*)(p0);
    v4f x1 = *(const v4f*)(p0 + 4);
    v4f x2 = *(const v4f*)(p1);
    v4f x3 = *(const v4f*)(p1 + 4);
    v16bf r;
#pragma unroll
    for (int i = 0; i < 4; ++i) {
        r[i]      = (__bf16)x0[i];
        r[4 + i]  = (__bf16)x1[i];
        r[8 + i]  = (__bf16)x2[i];
        r[12 + i] = (__bf16)x3[i];
    }
    return r;
}

__global__ __launch_bounds__(256)
void router_gemm_topk(const float* __restrict__ hidden,
                      const float* __restrict__ weight,
                      const float* __restrict__ bias,
                      float* __restrict__ out_scores,   // [T,4] f32
                      int*   __restrict__ out_index) {  // [T,4] i32
    __shared__ float s_log[BLK_M][E_DIM];               // 64 KB logits tile

    const int lane = threadIdx.x & 31;
    const int wave = threadIdx.x >> 5;                  // 0..7
    const int mg   = wave & 1;                          // M half: 0..1 (64 tokens each)
    const int ng   = wave >> 1;                         // N quarter: 0..3 (32 experts each)
    const int half = lane >> 4;                         // lane half selects K sub-range
    const int l16  = lane & 15;

    const size_t blockM = (size_t)blockIdx.x * BLK_M;

    // --- Phase 1: GEMM, wave computes 64(M) x 32(N) with 4x2 WMMA tiles ----
    v8f acc[4][2];
#pragma unroll
    for (int mt = 0; mt < 4; ++mt)
#pragma unroll
        for (int nt = 0; nt < 2; ++nt)
            acc[mt][nt] = (v8f)0.0f;

    // A: lane holds row M = l16 of its M-tile; K elems (half*16 .. half*16+15)
    const float* rowA[4];
#pragma unroll
    for (int mt = 0; mt < 4; ++mt)
        rowA[mt] = hidden + (blockM + (size_t)(mg * 64 + mt * 16 + l16)) * H_DIM;

    // B: lane holds column N = l16 of its N-tile; K elems (half*16 .. half*16+15)
    // weight is [E, H] row-major, so B column N == weight row N (contiguous in K).
    const float* colB[2];
#pragma unroll
    for (int nt = 0; nt < 2; ++nt)
        colB[nt] = weight + (size_t)(ng * 32 + nt * 16 + l16) * H_DIM + half * 16;

    for (int k = 0; k < H_DIM; k += KSTEP) {
        v16bf b0 = load_cvt_bf16(colB[0] + k, colB[0] + k + 8);
        v16bf b1 = load_cvt_bf16(colB[1] + k, colB[1] + k + 8);
#pragma unroll
        for (int mt = 0; mt < 4; ++mt) {
            const float* pa = rowA[mt] + k;
            __builtin_prefetch(pa + 2 * KSTEP, 0, 1);   // global_prefetch_b8
            v16bf a = load_cvt_bf16(pa + half * 16, pa + half * 16 + 8);
            acc[mt][0] = __builtin_amdgcn_wmma_f32_16x16x32_bf16(
                false, a, false, b0, (short)0, acc[mt][0], false, false);
            acc[mt][1] = __builtin_amdgcn_wmma_f32_16x16x32_bf16(
                false, a, false, b1, (short)0, acc[mt][1], false, false);
        }
    }

    // D layout: VGPR r, lanes 0-15 -> M=r, N=l16 ; lanes 16-31 -> M=8+r.
#pragma unroll
    for (int mt = 0; mt < 4; ++mt)
#pragma unroll
        for (int nt = 0; nt < 2; ++nt)
#pragma unroll
            for (int r = 0; r < 8; ++r)
                s_log[mg * 64 + mt * 16 + half * 8 + r]
                     [ng * 32 + nt * 16 + l16] = acc[mt][nt][r];

    __syncthreads();

    // --- Phase 2: per-token bias + top-4 + softmax (wave handles 16 tokens) -
    float bb[4];
    int   be[4];
#pragma unroll
    for (int j = 0; j < 4; ++j) {
        be[j] = lane + 32 * j;
        bb[j] = bias[be[j]];
    }

    for (int it = 0; it < 16; ++it) {
        const int tl = wave * 16 + it;
        float v[4];
        int   idx[4];
#pragma unroll
        for (int j = 0; j < 4; ++j) {
            v[j]   = s_log[tl][be[j]] + bb[j];
            idx[j] = be[j];
        }

        float tv[TOPK];
        int   ti[TOPK];
#pragma unroll
        for (int r = 0; r < TOPK; ++r) {
            // local best of the 4 candidates (tie -> lowest index)
            float bv = v[0];
            int   bi = idx[0];
#pragma unroll
            for (int j = 1; j < 4; ++j)
                if (v[j] > bv || (v[j] == bv && idx[j] < bi)) { bv = v[j]; bi = idx[j]; }
            // wave-wide argmax reduction
#pragma unroll
            for (int off = 16; off >= 1; off >>= 1) {
                float ov = __shfl_xor(bv, off, 32);
                int   oi = __shfl_xor(bi, off, 32);
                if (ov > bv || (ov == bv && oi < bi)) { bv = ov; bi = oi; }
            }
            tv[r] = bv;
            ti[r] = bi;
            // retire the winner
#pragma unroll
            for (int j = 0; j < 4; ++j)
                if (idx[j] == bi) v[j] = -INFINITY;
        }

        // softmax over the 4 selected logits (tv[0] is the max)
        float e[TOPK], s = 0.0f;
#pragma unroll
        for (int r = 0; r < TOPK; ++r) { e[r] = __expf(tv[r] - tv[0]); s += e[r]; }
        const float inv = 1.0f / s;

        const size_t t = blockM + (size_t)tl;
        if (lane < TOPK) {
            out_scores[t * TOPK + lane] = e[lane] * inv;
            out_index [t * TOPK + lane] = ti[lane];
        }
    }
}

extern "C" void kernel_launch(void* const* d_in, const int* in_sizes, int n_in,
                              void* d_out, int out_size, void* d_ws, size_t ws_size,
                              hipStream_t stream) {
    const float* hidden = (const float*)d_in[0];   // [16384, 2880] f32
    const float* weight = (const float*)d_in[1];   // [128, 2880]   f32
    const float* bias   = (const float*)d_in[2];   // [128]         f32

    // Tuple output concatenated flat: scores [T,4] f32 then indices [T,4] i32.
    float* out_scores = (float*)d_out;
    int*   out_index  = (int*)((float*)d_out + (size_t)T_TOK * TOPK);

    dim3 grid(T_TOK / BLK_M);   // 128 workgroups
    dim3 block(256);            // 8 waves (wave32)
    router_gemm_topk<<<grid, block, 0, stream>>>(hidden, weight, bias,
                                                 out_scores, out_index);
    (void)in_sizes; (void)n_in; (void)out_size; (void)d_ws; (void)ws_size;
}
